// CausalParticleAttention_20134806683953
// MI455X (gfx1250) — matrix operand order repaired and checked by reading, not compiled, not run on hardware
//
#include <hip/hip_runtime.h>
#include <hip/hip_bf16.h>

// ---------------- problem constants ----------------
#define Hn 8
#define Bn 16
#define Nn 16
#define Tn 48
#define Cn 512
#define HD 64
#define Ln (Nn * Tn)       // 768
#define Mrows (Bn * Ln)    // 12288
#define ATTN_WPB 4

typedef __attribute__((ext_vector_type(16))) _Float16 v16h;
typedef __attribute__((ext_vector_type(8)))  float    v8f;
typedef __attribute__((ext_vector_type(4)))  unsigned int u32x4;
typedef __attribute__((ext_vector_type(8)))  int      i32x8;
typedef __attribute__((ext_vector_type(4)))  int      i32x4;

// TDM path only on the device pass and only if the builtin exists there.
#if defined(__AMDGCN__) && __has_builtin(__builtin_amdgcn_tensor_load_to_lds)
#define USE_TDM 1
#else
#define USE_TDM 0
#endif

// ---------------- WMMA fragment loaders (CDNA5 16-bit layouts) ----------------
// A-matrix 16x32 f16: lane<16 holds row lane:   halves 0..7 = K+0..7,  8..15 = K+16..23
//                     lane>=16 holds row lane-16: halves 0..7 = K+8..15, 8..15 = K+24..31
__device__ __forceinline__ v16h load_a_frag(const _Float16* A, int lda, int m0, int k0, int lane) {
  int r    = lane & 15;
  int koff = (lane < 16) ? 0 : 8;
  const _Float16* p = A + (size_t)(m0 + r) * lda + k0 + koff;
  v16h v;
  ((float4*)&v)[0] = *(const float4*)(p);
  ((float4*)&v)[1] = *(const float4*)(p + 16);
  return v;
}

// B-matrix 32x16 f16 fed from column-major storage Bt[n][k]:
// lane<16 = column lane, halves 0..7 = K+0..7, 8..15 = K+16..23 ; lane>=16 = column lane-16, halves 0..7 = K+8..15, 8..15 = K+24..31
__device__ __forceinline__ v16h load_b_frag(const _Float16* Bt, int ldb, int n0, int k0, int lane) {
  int n    = lane & 15;
  int koff = (lane < 16) ? 0 : 8;
  const _Float16* p = Bt + (size_t)(n0 + n) * ldb + k0 + koff;
  v16h v;
  ((float4*)&v)[0] = *(const float4*)(p);
  ((float4*)&v)[1] = *(const float4*)(p + 16);
  return v;
}

__device__ __forceinline__ v8f wmma16(v16h a, v16h b, v8f c) {
  return __builtin_amdgcn_wmma_f32_16x16x32_f16(false, a, false, b, (short)0, c, false, false);
}

// ---------------- Tensor Data Mover: 2-D tile (rows x 32 f16) -> LDS ----------------
// D# per CDNA5 ISA. One DMA instruction per tile; TENSORcnt tracked.
__device__ __forceinline__ void tdm_load_2d(const _Float16* gsrc, int ld_elems, int rows,
                                            _Float16* ldst) {
#if USE_TDM
  unsigned long long ga = (unsigned long long)(uintptr_t)gsrc;
  unsigned int       la = (unsigned int)(uintptr_t)ldst;   // low 32 bits = LDS byte offset
  u32x4 g0;
  g0[0] = 1u;                                               // count=1, user mode
  g0[1] = la;                                               // lds_addr
  g0[2] = (unsigned int)ga;                                 // global_addr[31:0]
  g0[3] = (unsigned int)((ga >> 32) & 0x1FFFFFFu) | (2u << 30); // addr[56:32], type=2
  i32x8 g1;
  g1[0] = (int)(1u << 16);                 // workgroup_mask=0, data_size=1 (2 bytes)
  g1[1] = (int)0xFFFF0000u;                // tensor_dim0 lo16 (dim0 = 0x7FFFFFFF: no OOB clip)
  g1[2] = (int)0xFFFF7FFFu;                // dim0 hi16 | tensor_dim1 lo16 (dim1 huge)
  g1[3] = (int)((32u << 16) | 0x7FFFu);    // dim1 hi16 | tile_dim0 = 32 elems
  g1[4] = rows;                            // tile_dim1 = rows, tile_dim2 = 0
  g1[5] = ld_elems;                        // tensor_dim0_stride (elems), lo32
  g1[6] = 0;
  g1[7] = 0;
  i32x4 z4 = {0, 0, 0, 0};
#if __clang_major__ >= 23
  i32x8 z8 = {0, 0, 0, 0, 0, 0, 0, 0};
  __builtin_amdgcn_tensor_load_to_lds(g0, g1, z4, z4, z8, 0);
#else
  __builtin_amdgcn_tensor_load_to_lds(g0, g1, z4, z4, 0);
#endif
#endif
}

// Fallback cooperative copy (also used by the host-side parse when TDM is unavailable)
__device__ __forceinline__ void coop_copy_tile(const _Float16* g, int ld, int rows,
                                               _Float16* l, int tid) {
  int chunks = rows * 4;                 // 4 x 16B per 32-elem row
  for (int c = tid; c < chunks; c += 256) {
    int row = c >> 2, part = c & 3;
    *(float4*)(l + row * 32 + part * 8) = *(const float4*)(g + (size_t)row * ld + part * 8);
  }
}

// ---------------- conversion kernels ----------------
__global__ void cvt_f16_kernel(const float* __restrict__ in, _Float16* __restrict__ out, int n) {
  int i = blockIdx.x * blockDim.x + threadIdx.x;
  if (i < n) out[i] = (_Float16)in[i];
}

// W [K=512][N=512] f32 -> Wt [N][K] f16
__global__ void transpose512_f16_kernel(const float* __restrict__ W, _Float16* __restrict__ Wt) {
  int i = blockIdx.x * blockDim.x + threadIdx.x;
  int k = i >> 9, n = i & 511;
  Wt[((size_t)n << 9) + k] = (_Float16)W[((size_t)k << 9) + n];
}

// ---------------- TDM + LDS double-buffered WMMA GEMM ----------------
// Block = 8 waves, block tile 128x64, K-step 32. Wave w owns rows w*16 and the
// shared 64-column B tile (8x B-traffic reduction vs direct-global fragments).
// Epilogue modes:
//   0: fp32 store out[m*Cn + n]            (final Wo projection -> d_out)
//   1: f16 store [B][H][L][hd]             (Q, K)
//   2: f16 store [B][H][hd][L] (transposed, for V as the PV B-matrix)
__global__ __launch_bounds__(256) void wmma_gemm_kernel(
    const _Float16* __restrict__ A, const _Float16* __restrict__ Bt,
    int Ksz, float* __restrict__ outF, _Float16* __restrict__ outH, int mode) {
  __shared__ _Float16 ldsA[2][128 * 32];   // 2 x 8 KB
  __shared__ _Float16 ldsB[2][64 * 32];    // 2 x 4 KB

  int tid  = threadIdx.x;
  int lane = tid & 31;
  int w    = tid >> 5;
  int mblk = blockIdx.x * 128;
  int n0   = blockIdx.y * 64;
  const int KT = Ksz >> 5;

#if USE_TDM
  if (w == 0) {
    tdm_load_2d(A  + (size_t)mblk * Ksz, Ksz, 128, ldsA[0]);
    tdm_load_2d(Bt + (size_t)n0   * Ksz, Ksz,  64, ldsB[0]);
    if (KT > 1) {
      tdm_load_2d(A  + (size_t)mblk * Ksz + 32, Ksz, 128, ldsA[1]);
      tdm_load_2d(Bt + (size_t)n0   * Ksz + 32, Ksz,  64, ldsB[1]);
    }
  }
#endif

  v8f c0 = {}, c1 = {}, c2 = {}, c3 = {};
  for (int kt = 0; kt < KT; ++kt) {
    int buf = kt & 1;
#if USE_TDM
    if (w == 0) {
      // TDM completes in order per wave: <=2 outstanding ==> kt's two DMAs landed.
      if (kt + 1 < KT) __builtin_amdgcn_s_wait_tensorcnt(2);
      else             __builtin_amdgcn_s_wait_tensorcnt(0);
    }
    __syncthreads();
#else
    coop_copy_tile(A  + (size_t)mblk * Ksz + kt * 32, Ksz, 128, ldsA[buf], tid);
    coop_copy_tile(Bt + (size_t)n0   * Ksz + kt * 32, Ksz,  64, ldsB[buf], tid);
    __syncthreads();
#endif

    v16h a  = load_a_frag(ldsA[buf], 32, w * 16, 0, lane);
    v16h b0 = load_b_frag(ldsB[buf], 32, 0,  0, lane);
    v16h b1 = load_b_frag(ldsB[buf], 32, 16, 0, lane);
    v16h b2 = load_b_frag(ldsB[buf], 32, 32, 0, lane);
    v16h b3 = load_b_frag(ldsB[buf], 32, 48, 0, lane);
    c0 = wmma16(a, b0, c0);
    c1 = wmma16(a, b1, c1);
    c2 = wmma16(a, b2, c2);
    c3 = wmma16(a, b3, c3);

    __syncthreads();   // all reads of buf done before it is refilled
#if USE_TDM
    if (w == 0 && kt + 2 < KT) {
      tdm_load_2d(A  + (size_t)mblk * Ksz + (kt + 2) * 32, Ksz, 128, ldsA[buf]);
      tdm_load_2d(Bt + (size_t)n0   * Ksz + (kt + 2) * 32, Ksz,  64, ldsB[buf]);
    }
#endif
  }

  // C/D layout: lane<16 -> col=lane, rows 0..7 ; lane>=16 -> col=lane-16, rows 8..15
  int col   = lane & 15;
  int rbase = (lane < 16) ? 0 : 8;
  for (int r = 0; r < 8; ++r) {
    int m = mblk + w * 16 + rbase + r;
    float vals[4] = {c0[r], c1[r], c2[r], c3[r]};
    for (int j = 0; j < 4; ++j) {
      int n = n0 + j * 16 + col;
      if (mode == 0) {
        outF[(size_t)m * Cn + n] = vals[j];
      } else {
        int b = m / Ln, l = m % Ln, h = n / HD, d = n % HD;
        size_t bh = (size_t)(b * Hn + h);
        if (mode == 1) outH[(bh * Ln + l) * HD + d] = (_Float16)vals[j];
        else           outH[(bh * HD + d) * Ln + l] = (_Float16)vals[j];
      }
    }
  }
}

// ---------------- fused flash attention ----------------
// Q,K: [B][H][L][hd] f16 ; Vt: [B][H][hd][L] f16 ; Y: [B][L][C] f16
__global__ __launch_bounds__(ATTN_WPB * 32) void attn_kernel(
    const _Float16* __restrict__ Q, const _Float16* __restrict__ Kk,
    const _Float16* __restrict__ Vt, const float* __restrict__ btab,
    const float* __restrict__ ptab, _Float16* __restrict__ Y) {
  __shared__ _Float16 plds[ATTN_WPB][16 * 32];

  int lane  = threadIdx.x & 31;
  int w     = threadIdx.x >> 5;
  int qtile = blockIdx.x * ATTN_WPB + w;
  int q0    = qtile * 16;
  int bh    = blockIdx.y;
  int b     = bh / Hn, h = bh % Hn;

  const _Float16* Qbh = Q  + (size_t)bh * Ln * HD;
  const _Float16* Kbh = Kk + (size_t)bh * Ln * HD;
  const _Float16* Vbh = Vt + (size_t)bh * HD * Ln;

  // Q A-fragments for full hd=64 (two K-slices of 32)
  v16h aq0 = load_a_frag(Qbh, HD, q0, 0, lane);
  v16h aq1 = load_a_frag(Qbh, HD, q0, 32, lane);

  int col   = lane & 15;
  int rbase = (lane < 16) ? 0 : 8;
  int nq    = q0 / Tn;
  int tq0   = q0 % Tn;       // q-tile never crosses a particle (48 % 16 == 0)
  int tqmax = tq0 + 15;

  float mi[8], li[8];
  v8f acc[4];
  for (int r = 0; r < 8; ++r) { mi[r] = -3.0e38f; li[r] = 0.0f; }
  v8f zero = {};
  for (int db = 0; db < 4; ++db) acc[db] = zero;

  const float scale = 0.125f;  // 1/sqrt(64)

  for (int kb = 0; kb < Ln / 32; ++kb) {
    int kstart = kb * 32;
    // causal skip: min t over this key block > max t of query tile -> fully masked
    int tstart = kstart % Tn;
    int mint   = (tstart + 31 >= Tn) ? 0 : tstart;
    if (mint > tqmax) continue;

    // S tile 16x32 = Q(16x64) . K^T, two 16-key halves
    v8f s0 = {}, s1 = {};
    {
      v16h bk = load_b_frag(Kbh, HD, kstart, 0, lane);
      s0 = wmma16(aq0, bk, s0);
      bk = load_b_frag(Kbh, HD, kstart, 32, lane);
      s0 = wmma16(aq1, bk, s0);
      bk = load_b_frag(Kbh, HD, kstart + 16, 0, lane);
      s1 = wmma16(aq0, bk, s1);
      bk = load_b_frag(Kbh, HD, kstart + 16, 32, lane);
      s1 = wmma16(aq1, bk, s1);
    }

    int kk0 = kstart + col,  kk1 = kstart + 16 + col;
    int tk0 = kk0 % Tn, nk0 = kk0 / Tn;
    int tk1 = kk1 % Tn, nk1 = kk1 / Tn;
    float bp0 = ptab[(nk0 - nq + Nn - 1) * Hn + h];
    float bp1 = ptab[(nk1 - nq + Nn - 1) * Hn + h];

    float p0[8], p1[8];
    for (int r = 0; r < 8; ++r) {
      int tq = tq0 + rbase + r;
      float v0 = s0[r] * scale + btab[(tk0 - tq + Tn - 1) * Hn + h] + bp0;
      float v1 = s1[r] * scale + btab[(tk1 - tq + Tn - 1) * Hn + h] + bp1;
      if (tk0 > tq) v0 = -1.0e30f;
      if (tk1 > tq) v1 = -1.0e30f;
      // each S row lives in one VGPR across a 16-lane half -> shfl_xor reduction
      float vmax = fmaxf(v0, v1);
      for (int off = 1; off < 16; off <<= 1) vmax = fmaxf(vmax, __shfl_xor(vmax, off, 32));
      float mn = fmaxf(mi[r], vmax);
      float e0 = __expf(v0 - mn);
      float e1 = __expf(v1 - mn);
      p0[r] = e0; p1[r] = e1;
      float rs = e0 + e1;
      for (int off = 1; off < 16; off <<= 1) rs += __shfl_xor(rs, off, 32);
      float alpha = __expf(mi[r] - mn);
      li[r] = li[r] * alpha + rs;
      mi[r] = mn;
      for (int db = 0; db < 4; ++db) acc[db][r] *= alpha;
    }

    // transpose P (C-layout) -> A-fragment layout via per-wave LDS tile.
    // No block barrier (waves diverge on the causal skip); LDS ops are in-order
    // per wave, so s_wait_dscnt 0 between stores and loads suffices.
    _Float16* tile = plds[w];
    for (int r = 0; r < 8; ++r) {
      int row = rbase + r;
      tile[row * 32 + col]      = (_Float16)p0[r];
      tile[row * 32 + 16 + col] = (_Float16)p1[r];
    }
    asm volatile("s_wait_dscnt 0x0" ::: "memory");
    v16h pa;
    {
      int row  = lane & 15;
      int koff = (lane < 16) ? 0 : 8;
      const _Float16* tp = tile + row * 32 + koff;
      ((float4*)&pa)[0] = *(const float4*)(tp);
      ((float4*)&pa)[1] = *(const float4*)(tp + 16);
    }

    // O += P(16x32) . V(32x64) : V columns contiguous thanks to Vt layout
    for (int db = 0; db < 4; ++db) {
      v16h bv = load_b_frag(Vbh, Ln, db * 16, kstart, lane);
      acc[db] = wmma16(pa, bv, acc[db]);
    }
  }

  // normalize and scatter to [B][L][C] f16 for the Wo GEMM
  for (int r = 0; r < 8; ++r) {
    float inv = 1.0f / li[r];
    int l = q0 + rbase + r;
    for (int db = 0; db < 4; ++db) {
      int c = h * HD + db * 16 + col;
      Y[((size_t)b * Ln + l) * Cn + c] = (_Float16)(acc[db][r] * inv);
    }
  }
}

// ---------------- host launcher ----------------
extern "C" void kernel_launch(void* const* d_in, const int* in_sizes, int n_in,
                              void* d_out, int out_size, void* d_ws, size_t ws_size,
                              hipStream_t stream) {
  const float* x  = (const float*)d_in[0];
  const float* Wq = (const float*)d_in[1];
  const float* Wk = (const float*)d_in[2];
  const float* Wv = (const float*)d_in[3];
  const float* Wo = (const float*)d_in[4];
  const float* bt = (const float*)d_in[5];
  const float* bp = (const float*)d_in[6];
  float* out = (float*)d_out;

  char* ws = (char*)d_ws;
  size_t off = 0;
  auto alloc = [&](size_t bytes) {
    char* p = ws + off;
    off = (off + bytes + 255) & ~(size_t)255;
    return p;
  };
  _Float16* x16 = (_Float16*)alloc((size_t)Mrows * Cn * 2);
  _Float16* Wqt = (_Float16*)alloc((size_t)Cn * Cn * 2);
  _Float16* Wkt = (_Float16*)alloc((size_t)Cn * Cn * 2);
  _Float16* Wvt = (_Float16*)alloc((size_t)Cn * Cn * 2);
  _Float16* Wot = (_Float16*)alloc((size_t)Cn * Cn * 2);
  _Float16* Q16 = (_Float16*)alloc((size_t)Bn * Hn * Ln * HD * 2);
  _Float16* K16 = (_Float16*)alloc((size_t)Bn * Hn * Ln * HD * 2);
  _Float16* V16 = (_Float16*)alloc((size_t)Bn * Hn * Ln * HD * 2);
  _Float16* Y16 = (_Float16*)alloc((size_t)Mrows * Cn * 2);

  // precision conversion + weight transposes
  int nx = Mrows * Cn;
  cvt_f16_kernel<<<(nx + 255) / 256, 256, 0, stream>>>(x, x16, nx);
  transpose512_f16_kernel<<<(Cn * Cn) / 256, 256, 0, stream>>>(Wq, Wqt);
  transpose512_f16_kernel<<<(Cn * Cn) / 256, 256, 0, stream>>>(Wk, Wkt);
  transpose512_f16_kernel<<<(Cn * Cn) / 256, 256, 0, stream>>>(Wv, Wvt);
  transpose512_f16_kernel<<<(Cn * Cn) / 256, 256, 0, stream>>>(Wo, Wot);

  // QKV projections (WMMA + TDM staging), V stored transposed for the PV matmul
  dim3 gg(Mrows / 128, Cn / 64);
  wmma_gemm_kernel<<<gg, 256, 0, stream>>>(x16, Wqt, Cn, nullptr, Q16, 1);
  wmma_gemm_kernel<<<gg, 256, 0, stream>>>(x16, Wkt, Cn, nullptr, K16, 1);
  wmma_gemm_kernel<<<gg, 256, 0, stream>>>(x16, Wvt, Cn, nullptr, V16, 2);

  // fused causal attention with biases + online softmax
  dim3 ag(Ln / 16 / ATTN_WPB, Bn * Hn);
  attn_kernel<<<ag, ATTN_WPB * 32, 0, stream>>>(Q16, K16, V16, bt, bp, Y16);

  // output projection, fp32 result
  wmma_gemm_kernel<<<gg, 256, 0, stream>>>(Y16, Wot, Cn, out, nullptr, 0);
}
